// kernel_generated_1_62904091017272
// MI455X (gfx1250) — matrix operand order repaired and checked by reading, not compiled, not run on hardware
//
#include <hip/hip_runtime.h>
#include <stdint.h>

// ---------------------------------------------------------------------------
// Implicit-GEMM shifted-window grouped conv for MI455X (gfx1250, wave32).
//   out[m,l,h,o] = sum_{j,i,k} A[(m,h,o),(j,i,k)] * w[j,i,k,l]
// GEMM: Mdim = l (128), Ndim = (m,h,o) (100352), K = 192, via
// V_WMMA_F32_16X16X32_BF16 (bf16 in, f32 accumulate).
// ---------------------------------------------------------------------------

typedef __attribute__((ext_vector_type(16))) __bf16 v16bf;
typedef __attribute__((ext_vector_type(8)))  float  v8f;

union FragU { uint4 q[2]; v16bf v; };  // 32 bytes either way

__device__ __forceinline__ unsigned short f2bf(float f) {
  // round-to-nearest-even f32 -> bf16
  unsigned int u = __float_as_uint(f);
  u += 0x7fffu + ((u >> 16) & 1u);
  return (unsigned short)(u >> 16);
}

#define KDIM 192   // reduction: (j,i,k) = 2*32*3
#define KSTR 200   // padded LDS row stride in halves (multiple of 8 -> 16B aligned)
#define ROWS 64    // spatial rows per workgroup (4 waves x 16)
#define NPOS 784   // 28*28 spatial positions per image
#define NL   128   // output channels

// Transform w[2][32][3][128] f32 -> WT[l][r] bf16, r = (j*32+i)*3+k.
__global__ void prep_w_kernel(const float* __restrict__ w,
                              unsigned short* __restrict__ wt) {
  int idx = blockIdx.x * blockDim.x + threadIdx.x;
  if (idx < KDIM * NL) {
    int l = idx / KDIM;
    int r = idx - l * KDIM;
    wt[idx] = f2bf(w[r * NL + l]);   // w flat index = r*128 + l
  }
}

__global__ void __launch_bounds__(128)
conv_shift_wmma(const float* __restrict__ x,
                const unsigned short* __restrict__ wt,
                float* __restrict__ out) {
  // bf16 patch tile: 64 spatial rows x 192 K, padded stride vs LDS banks
  __shared__ unsigned short Pt[ROWS * KSTR];   // 25600 B

  const int t      = threadIdx.x;      // 0..127
  const int wave   = t >> 5;
  const int lane   = t & 31;
  const int lh     = lane & 15;
  const bool hi    = lane >= 16;
  const int wgBase = blockIdx.x * ROWS;

  // ---- cooperative gather of the patch tile (implicit im2col) ------------
  // thread t: spatial row = t>>1, j = t&1 covers 96 columns (i=0..31, k=0..2)
  {
    const int row = t >> 1;
    const int j   = t & 1;
    const int Rg  = wgBase + row;           // global spatial row (m,h,o)
    const int m   = Rg / NPOS;
    const int rem = Rg - m * NPOS;
    const int h   = rem / 28;
    const int o   = rem - h * 28;
    // j==1 -> channels 0..31, row h ; j==0 -> channels 32..63, row (h-1) mod 28
    const int hr = j ? h : (h + 27) % 28;
    const int cb = j ? 0 : 32;
    const float* xr = x + ((m * 64 + cb) * 28 + hr) * 28;  // + i*784 + col
    unsigned short* pr = Pt + row * KSTR + j * 96;
    #pragma unroll
    for (int k = 0; k < 3; ++k) {
      const int  w0    = o + k - 1;
      const bool valid = (unsigned)w0 < 28u;     // zero-pad gate
      const int  col   = (w0 + 27) % 28;         // (w0-1) mod 28 shift
      #pragma unroll 8
      for (int i = 0; i < 32; ++i) {
        float v = valid ? xr[i * NPOS + col] : 0.0f;
        pr[i * 3 + k] = f2bf(v);
      }
    }
  }
  __syncthreads();

  // ---- WMMA main loop ----------------------------------------------------
  v8f acc[8];
  #pragma unroll
  for (int i = 0; i < 8; ++i)
    acc[i] = (v8f){0.f, 0.f, 0.f, 0.f, 0.f, 0.f, 0.f, 0.f};

  // B frag (patch, KxN): lane holds col N = wave*16+lh, K runs [Kb,Kb+8) and [Kb+16,Kb+24), Kb = 0/8
  const unsigned short* Bbase = Pt + (wave * 16 + lh) * KSTR + (hi ? 8 : 0);
  // A frag (weights, MxK): lane holds row M(=l) = mt*16+lh,
  // K runs [Kb,Kb+8) and [Kb+16,Kb+24), Kb = 0/8. Read straight from L2-resident WT.
  const unsigned short* Abase = wt + lh * KDIM + (hi ? 8 : 0);

  #pragma unroll
  for (int kc = 0; kc < 6; ++kc) {             // K = 6 x 32
    FragU bf;
    bf.q[0] = *(const uint4*)(Bbase + kc * 32);
    bf.q[1] = *(const uint4*)(Bbase + kc * 32 + 16);
    #pragma unroll
    for (int mt = 0; mt < 8; ++mt) {           // 8 l-tiles of 16
      const unsigned short* ap = Abase + mt * 16 * KDIM + kc * 32;
      FragU af;
      af.q[0] = *(const uint4*)ap;             // K [Kb, Kb+8)
      af.q[1] = *(const uint4*)(ap + 16);      // K [Kb+16, Kb+24)
      acc[mt] = __builtin_amdgcn_wmma_f32_16x16x32_bf16(
          false, af.v, false, bf.v, (short)0, acc[mt], false, false);
    }
  }

  // ---- store: N (spatial) across lanes -> coalesced 64B segments ---------
  // D layout: VGPR r -> M = r + (hi?8:0), N = lh. Tile never crosses m (784%16==0).
  const int Rt = wgBase + wave * 16;
  const int m  = Rt / NPOS;
  const int p  = Rt - m * NPOS;
  float* ob = out + (size_t)m * NL * NPOS + p + lh;
  const int lofs = hi ? 8 : 0;
  #pragma unroll
  for (int mt = 0; mt < 8; ++mt) {
    #pragma unroll
    for (int r = 0; r < 8; ++r) {
      const int l = mt * 16 + lofs + r;
      ob[(size_t)l * NPOS] = acc[mt][r];
    }
  }
}

extern "C" void kernel_launch(void* const* d_in, const int* in_sizes, int n_in,
                              void* d_out, int out_size, void* d_ws, size_t ws_size,
                              hipStream_t stream) {
  const float* x = (const float*)d_in[0];   // [128,64,28,28] f32
  const float* w = (const float*)d_in[1];   // [2,32,3,128]   f32
  float* out = (float*)d_out;               // [128,128,28,28] f32
  unsigned short* wt = (unsigned short*)d_ws;  // 192*128 bf16 = 48 KB scratch

  prep_w_kernel<<<(KDIM * NL + 255) / 256, 256, 0, stream>>>(w, wt);
  conv_shift_wmma<<<(NL * NPOS) / ROWS, 128, 0, stream>>>(x, wt, out);
}
